// PowerSoftMin_67413806678292
// MI455X (gfx1250) — matrix operand order repaired and checked by reading, not compiled, not run on hardware
//
#include <hip/hip_runtime.h>
#include <hip/hip_bf16.h>
#include <math.h>

// Problem constants (from reference): B=4, N=M=4096, D_FEAT=64, T=0.01, POWER=3
#define BATCH   4
#define NPTS    4096
#define DFEAT   64
// base-2 logits: 2^(-C2*d) == e^(-100*d), C2 = 100*log2(e)
#define C2_TEMP 144.26950408889634f

typedef __attribute__((ext_vector_type(16))) __bf16 v16bf;
typedef __attribute__((ext_vector_type(8)))  __bf16 v8bf;
typedef __attribute__((ext_vector_type(8)))  float  v8f;

// ---------------------------------------------------------------------------
// Pass 0: split f32 points into bf16 (hi, lo) pairs and compute squared norms.
// x = hi + lo with hi = bf16(x), lo = bf16(x - hi)  ->  ~2^-17 relative error
// when recombined as hi*hi + hi*lo + lo*hi in f32 WMMA accumulators.
// ---------------------------------------------------------------------------
__global__ void prep_split(const float* __restrict__ src,
                           __bf16* __restrict__ hi, __bf16* __restrict__ lo,
                           float* __restrict__ nrm, int nrows)
{
    int r = blockIdx.x * blockDim.x + threadIdx.x;
    if (r >= nrows) return;
    const float* p  = src + (size_t)r * DFEAT;
    __bf16*      ph = hi  + (size_t)r * DFEAT;
    __bf16*      pl = lo  + (size_t)r * DFEAT;
    float s = 0.0f;
#pragma unroll 8
    for (int k = 0; k < DFEAT; ++k) {
        float v = p[k];
        s = fmaf(v, v, s);
        __bf16 h = (__bf16)v;
        ph[k] = h;
        pl[k] = (__bf16)(v - (float)h);
    }
    nrm[r] = s;
}

// A-fragment loader: contiguous K order (matches B fragment).
// lane half 0: elems 0..15 = K k0+0..15
// lane half 1: elems 0..15 = K k0+16..31
__device__ inline v16bf load_a16(const __bf16* rowbase, int k0, int half)
{
    const v8bf e0 = *(const v8bf*)(rowbase + k0 + 16 * half);
    const v8bf e1 = *(const v8bf*)(rowbase + k0 + 16 * half + 8);
    v16bf r;
#pragma unroll
    for (int i = 0; i < 8; ++i) { r[i] = e0[i]; r[i + 8] = e1[i]; }
    return r;
}

// ---------------------------------------------------------------------------
// out[b, n] = sum_m softmax_m(-100*D[n,m]) * D[n,m]  (row-axis softmin).
// Called twice with (x,y) and (y,x) swapped since D(x,y)^T = D(y,x).
// One wave owns TWO 16-row tiles (32 rows) so every B fragment loaded from L2
// feeds 12 WMMAs; online-softmin state (base-2 max, sumexp, sumexp*d) lives
// in VGPRs per C-slot and is merged across the 16 lanes of each half-wave
// only once at the end.
// ---------------------------------------------------------------------------
__global__ __launch_bounds__(128) void softmin_rows(
    const __bf16* __restrict__ Ahi, const __bf16* __restrict__ Alo,
    const float*  __restrict__ A2,
    const __bf16* __restrict__ Bhi, const __bf16* __restrict__ Blo,
    const float*  __restrict__ B2,
    float* __restrict__ outRow)
{
    const int lane = threadIdx.x & 31;
    const int wave = threadIdx.x >> 5;
    const int half = lane >> 4;       // which 16-lane half of the wave32
    const int l16  = lane & 15;

    const int tile          = blockIdx.x * 4 + wave;   // 4 waves / block
    const int tilesPerBatch = NPTS / 32;               // 32 rows per wave
    const int b             = tile / tilesPerBatch;
    const int n0            = (tile % tilesPerBatch) * 32;

    const size_t matOff = (size_t)b * NPTS * DFEAT;
    const __bf16* aHi = Ahi + matOff;
    const __bf16* aLo = Alo + matOff;
    const __bf16* bHi = Bhi + matOff;
    const __bf16* bLo = Blo + matOff;
    const float*  a2  = A2 + (size_t)b * NPTS;
    const float*  b2  = B2 + (size_t)b * NPTS;
    float*        op  = outRow + (size_t)b * NPTS;

    // Hold both 16-row A tiles (hi & lo, both K-halves) in VGPRs.
    v16bf a_h0[2], a_h1[2], a_l0[2], a_l1[2];
#pragma unroll
    for (int t = 0; t < 2; ++t) {
        const __bf16* arow_h = aHi + (size_t)(n0 + 16 * t + l16) * DFEAT;
        const __bf16* arow_l = aLo + (size_t)(n0 + 16 * t + l16) * DFEAT;
        a_h0[t] = load_a16(arow_h, 0, half);
        a_h1[t] = load_a16(arow_h, 32, half);
        a_l0[t] = load_a16(arow_l, 0, half);
        a_l1[t] = load_a16(arow_l, 32, half);
    }

    // Row norms: C VGPR v of tile t holds row n0 + 16t + v + 8*half.
    float x2row[2][8];
#pragma unroll
    for (int t = 0; t < 2; ++t)
#pragma unroll
        for (int v = 0; v < 8; ++v)
            x2row[t][v] = a2[n0 + 16 * t + v + 8 * half];

    float mrun[2][8], srun[2][8], wrun[2][8];
#pragma unroll
    for (int t = 0; t < 2; ++t)
#pragma unroll
        for (int v = 0; v < 8; ++v) {
            mrun[t][v] = -__builtin_inff(); srun[t][v] = 0.0f; wrun[t][v] = 0.0f;
        }

    for (int m0 = 0; m0 < NPTS; m0 += 16) {
        // B-fragment: contiguous K order (same as A) -> 16 contiguous bf16 at
        // col*DFEAT + k0 + 16*half (32-byte aligned load).
        const __bf16* bcol_h = bHi + (size_t)(m0 + l16) * DFEAT + 16 * half;
        const __bf16* bcol_l = bLo + (size_t)(m0 + l16) * DFEAT + 16 * half;
        const v16bf b_h0 = *(const v16bf*)(bcol_h);
        const v16bf b_h1 = *(const v16bf*)(bcol_h + 32);
        const v16bf b_l0 = *(const v16bf*)(bcol_l);
        const v16bf b_l1 = *(const v16bf*)(bcol_l + 32);
        const float y2c  = b2[m0 + l16];

#pragma unroll
        for (int t = 0; t < 2; ++t) {
            // dot ~= hi.hi + hi.lo + lo.hi over K=64 (two K=32 steps): 6 WMMAs.
            v8f acc = {};
            acc = __builtin_amdgcn_wmma_f32_16x16x32_bf16(false, a_h0[t], false, b_h0, (short)0, acc, false, false);
            acc = __builtin_amdgcn_wmma_f32_16x16x32_bf16(false, a_h1[t], false, b_h1, (short)0, acc, false, false);
            acc = __builtin_amdgcn_wmma_f32_16x16x32_bf16(false, a_h0[t], false, b_l0, (short)0, acc, false, false);
            acc = __builtin_amdgcn_wmma_f32_16x16x32_bf16(false, a_h1[t], false, b_l1, (short)0, acc, false, false);
            acc = __builtin_amdgcn_wmma_f32_16x16x32_bf16(false, a_l0[t], false, b_h0, (short)0, acc, false, false);
            acc = __builtin_amdgcn_wmma_f32_16x16x32_bf16(false, a_l1[t], false, b_h1, (short)0, acc, false, false);

#pragma unroll
            for (int v = 0; v < 8; ++v) {
                float d2 = fmaf(-2.0f, acc[v], x2row[t][v] + y2c);
                float d  = __builtin_amdgcn_sqrtf(fmaxf(d2, 0.0f)); // raw v_sqrt_f32
                float lg = -C2_TEMP * d;                            // base-2 logit
                // Branchless online softmax, one bare v_exp_f32 per element:
                float mo = mrun[t][v];
                float mn = fmaxf(mo, lg);
                float e  = __builtin_amdgcn_exp2f(fminf(mo, lg) - mn);
                float p  = (mo >= lg) ? 1.0f : e;   // rescale of old accum
                float q  = (mo >= lg) ? e : 1.0f;   // weight of new element
                srun[t][v] = fmaf(srun[t][v], p, q);
                wrun[t][v] = fmaf(wrun[t][v], p, q * d);
                mrun[t][v] = mn;
            }
        }
    }

    // Merge the 16 per-lane partial softmin states per row (butterfly within
    // each half-wave; XOR masks 1,2,4,8 stay inside the half).
#pragma unroll
    for (int t = 0; t < 2; ++t)
#pragma unroll
        for (int v = 0; v < 8; ++v) {
            float m = mrun[t][v], s = srun[t][v], w = wrun[t][v];
#pragma unroll
            for (int off = 1; off < 16; off <<= 1) {
                float m2 = __shfl_xor(m, off, 32);
                float s2 = __shfl_xor(s, off, 32);
                float w2 = __shfl_xor(w, off, 32);
                float mn = fmaxf(m, m2);
                float e  = __builtin_amdgcn_exp2f(fminf(m, m2) - mn);
                bool keep = (m >= m2);
                s = keep ? fmaf(s2, e, s) : fmaf(s, e, s2);
                w = keep ? fmaf(w2, e, w) : fmaf(w, e, w2);
                m = mn;
            }
            if (l16 == 0) op[n0 + 16 * t + v + 8 * half] = w / s;
        }
}

// ---------------------------------------------------------------------------
// loss = ( sum sm_prec^3 + sum sm_cov^3 ) / (B*N)   (since N == M)
// ---------------------------------------------------------------------------
__global__ void finalize_loss(const float* __restrict__ smPrec,
                              const float* __restrict__ smCov,
                              float* __restrict__ out, int nPerArr)
{
    __shared__ float red[256];
    float acc = 0.0f;
    for (int i = threadIdx.x; i < nPerArr; i += 256) {
        float a = smPrec[i], c = smCov[i];
        acc += a * a * a + c * c * c;
    }
    red[threadIdx.x] = acc;
    __syncthreads();
    for (int s = 128; s > 0; s >>= 1) {
        if ((int)threadIdx.x < s) red[threadIdx.x] += red[threadIdx.x + s];
        __syncthreads();
    }
    if (threadIdx.x == 0) out[0] = red[0] / (float)nPerArr;
}

extern "C" void kernel_launch(void* const* d_in, const int* in_sizes, int n_in,
                              void* d_out, int out_size, void* d_ws, size_t ws_size,
                              hipStream_t stream)
{
    const float* x = (const float*)d_in[0];   // [B, N, 64] f32
    const float* y = (const float*)d_in[1];   // [B, M, 64] f32
    float* out = (float*)d_out;               // scalar f32

    char* ws = (char*)d_ws;
    const size_t rows     = (size_t)BATCH * NPTS;               // 16384
    const size_t matBytes = rows * DFEAT * sizeof(__bf16);      // 2 MiB each
    __bf16* xhi = (__bf16*)ws; ws += matBytes;
    __bf16* xlo = (__bf16*)ws; ws += matBytes;
    __bf16* yhi = (__bf16*)ws; ws += matBytes;
    __bf16* ylo = (__bf16*)ws; ws += matBytes;
    float* x2     = (float*)ws; ws += rows * sizeof(float);
    float* y2     = (float*)ws; ws += rows * sizeof(float);
    float* smPrec = (float*)ws; ws += rows * sizeof(float);
    float* smCov  = (float*)ws; ws += rows * sizeof(float);

    dim3 pB(256), pG((unsigned)((rows + 255) / 256));
    prep_split<<<pG, pB, 0, stream>>>(x, xhi, xlo, x2, (int)rows);
    prep_split<<<pG, pB, 0, stream>>>(y, yhi, ylo, y2, (int)rows);

    const int tiles = BATCH * (NPTS / 32);    // 512 waves, 4 waves/block
    dim3 mB(128), mG(tiles / 4);
    // precision: softmin over targets per prediction row
    softmin_rows<<<mG, mB, 0, stream>>>(xhi, xlo, x2, yhi, ylo, y2, smPrec);
    // coverage: same computation on the transposed problem
    softmin_rows<<<mG, mB, 0, stream>>>(yhi, ylo, y2, xhi, xlo, x2, smCov);

    finalize_loss<<<1, 256, 0, stream>>>(smPrec, smCov, out, (int)rows);
}
